// RoutedSquaredReLUReadout_43550968382274
// MI455X (gfx1250) — compile-verified, matrix operand order checked by reading, not run on hardware
//
#include <hip/hip_runtime.h>

#define DD 1024
#define HH 2048
#define OO 1024
#define EE 8
#define NN 4096

// LDS tile: 128 rows x 32 bf16 (64B/row), padded to 80B/row to spread banks.
#define ROWE 40                   // row stride in elements (80 bytes)
#define TILE_LDS (128 * ROWE)     // elements per tile buffer
#define TILE_B  ((unsigned)(TILE_LDS * 2))   // bytes per tile buffer

typedef __attribute__((ext_vector_type(16))) __bf16        v16bf;
typedef __attribute__((ext_vector_type(8)))  float         v8f;
typedef __attribute__((ext_vector_type(4)))  float         v4f;
typedef __attribute__((ext_vector_type(4)))  unsigned int  v4u;
typedef __attribute__((ext_vector_type(2)))  unsigned int  v2u;
typedef unsigned short u16;
typedef unsigned int   u32;

union FragBF16 { v16bf v; v4u q[2]; };

__device__ __forceinline__ u16 f2bf(float f) {
    u32 u = __float_as_uint(f);
    u += 0x7fffu + ((u >> 16) & 1u);   // round-to-nearest-even
    return (u16)(u >> 16);
}

__device__ __forceinline__ unsigned lds_off_u32(const void* p) {
    // Generic pointer to a __shared__ object: low 32 bits are the LDS byte offset.
    // Kept as a scalar (never stored in an aggregate) so no addrspacecast constant
    // ever lands in a static initializer.
    return (unsigned)(size_t)p;
}

// Async DMA: 16 bytes global -> LDS, GVS addressing (saddr base + 32-bit voffset).
// Tracked by ASYNCcnt; completes in issue order.
__device__ __forceinline__ void async_b128(unsigned ldsoff, unsigned voff, const void* sbase) {
    asm volatile("global_load_async_to_lds_b128 %0, %1, %2"
                 :: "v"(ldsoff), "v"(voff), "s"(sbase)
                 : "memory");
}

template <int N>
__device__ __forceinline__ void wait_async() {
#if __has_builtin(__builtin_amdgcn_s_wait_asynccnt)
    __builtin_amdgcn_s_wait_asynccnt(N);
#else
    asm volatile("s_wait_asynccnt %0" :: "i"(N) : "memory");
#endif
}

// Copy one 128x32-bf16 A tile and one B tile into LDS (512 b128 transfers each,
// 4 per thread per tile with 128 threads). ldbytes = global K-stride in bytes.
__device__ __forceinline__ void issue_tile_pair(const char* gA, const char* gB,
                                                unsigned ldbytes,
                                                unsigned lA, unsigned lB, int tid) {
#pragma unroll
    for (int j = 0; j < 4; ++j) {
        const int idx = tid + 128 * j;
        const int row = idx >> 2;
        const int ch  = idx & 3;
        const unsigned voff = (unsigned)row * ldbytes + (unsigned)ch * 16u;
        const unsigned lofs = (unsigned)row * (ROWE * 2) + (unsigned)ch * 16u;
        async_b128(lA + lofs, voff, gA);
    }
#pragma unroll
    for (int j = 0; j < 4; ++j) {
        const int idx = tid + 128 * j;
        const int row = idx >> 2;
        const int ch  = idx & 3;
        const unsigned voff = (unsigned)row * ldbytes + (unsigned)ch * 16u;
        const unsigned lofs = (unsigned)row * (ROWE * 2) + (unsigned)ch * 16u;
        async_b128(lB + lofs, voff, gB);
    }
}

// Load one 16x32 bf16 WMMA A/B fragment from an LDS tile (row-major, ROWE stride).
// ISA 16-bit A/B layout: lanes 0-15 hold K{0..7,16..23} of row (lane&15),
// lanes 16-31 hold K{8..15,24..31}. Two ds_load_b128 per lane.
__device__ __forceinline__ v16bf load_frag_lds(const u16* p0, int lane) {
    const int rsel = lane & 15;
    const int hi   = lane >> 4;
    const u16* p = p0 + rsel * ROWE + hi * 8;
    FragBF16 f;
    f.q[0] = *reinterpret_cast<const v4u*>(p);
    f.q[1] = *reinterpret_cast<const v4u*>(p + 16);
    return f.v;
}

// ---------------------------------------------------------------------------
// f32 -> bf16 bulk convert (b128 load, b64 store)
// ---------------------------------------------------------------------------
__global__ __launch_bounds__(256) void convert_kernel(const float* __restrict__ src,
                                                      u16* __restrict__ dst, int n4) {
    int i = blockIdx.x * blockDim.x + threadIdx.x;
    if (i >= n4) return;
    v4f v = reinterpret_cast<const v4f*>(src)[i];
    v2u o;
    o.x = (u32)f2bf(v.x) | ((u32)f2bf(v.y) << 16);
    o.y = (u32)f2bf(v.z) | ((u32)f2bf(v.w) << 16);
    reinterpret_cast<v2u*>(dst)[i] = o;
}

// ---------------------------------------------------------------------------
// Router: route[n,e] = softmax_e(x[n,:] @ Wr[e,:] + br[e]).  One wave per row.
// ---------------------------------------------------------------------------
__global__ __launch_bounds__(256) void router_kernel(const float* __restrict__ x,
                                                     const float* __restrict__ Wr,
                                                     const float* __restrict__ br,
                                                     float* __restrict__ route) {
    const int lane = threadIdx.x & 31;
    const int wid  = threadIdx.x >> 5;
    const int n    = blockIdx.x * 8 + wid;

    float acc[EE];
#pragma unroll
    for (int e = 0; e < EE; ++e) acc[e] = 0.0f;

    const float* xr = x + (size_t)n * DD;
    for (int d = lane; d < DD; d += 32) {
        const float xv = xr[d];
#pragma unroll
        for (int e = 0; e < EE; ++e) acc[e] += xv * Wr[(size_t)e * DD + d];
    }
#pragma unroll
    for (int e = 0; e < EE; ++e) {
#pragma unroll
        for (int off = 16; off > 0; off >>= 1)
            acc[e] += __shfl_xor(acc[e], off, 32);
    }
    if (lane == 0) {
        float m = -3.4e38f;
#pragma unroll
        for (int e = 0; e < EE; ++e) { acc[e] += br[e]; m = fmaxf(m, acc[e]); }
        float s = 0.0f;
#pragma unroll
        for (int e = 0; e < EE; ++e) { acc[e] = __expf(acc[e] - m); s += acc[e]; }
        const float inv = 1.0f / s;
#pragma unroll
        for (int e = 0; e < EE; ++e) route[(size_t)n * EE + e] = acc[e] * inv;
    }
}

// ---------------------------------------------------------------------------
// GEMM1: hsq[e,n,h] = sqrelu(x[n,:] @ W1[e,h,:] + b1[e,h])  (bf16 out)
// Block = 4 waves (2x2 of 64x64 wave tiles) = 128x128 tile; K-step 32.
// A/B tiles staged in LDS via async copies, double-buffered.
// ---------------------------------------------------------------------------
__global__ __launch_bounds__(128) void gemm1_kernel(const u16* __restrict__ xbf,
                                                    const u16* __restrict__ w1bf,
                                                    const float* __restrict__ b1,
                                                    u16* __restrict__ hsq) {
    __shared__ __align__(16) u16 smem[4 * TILE_LDS];

    const int tid  = threadIdx.x;
    const int lane = tid & 31;
    const int wid  = tid >> 5;
    const int e    = blockIdx.z;
    const int mb   = blockIdx.y * 128;
    const int nb   = blockIdx.x * 128;
    const int wm   = (wid >> 1) * 64;
    const int wn   = (wid & 1) * 64;

    const char* gA = (const char*)(xbf + (size_t)mb * DD);
    const char* gB = (const char*)(w1bf + ((size_t)e * HH + nb) * DD);

    const unsigned lbase = lds_off_u32(smem);   // scalar only; never in an aggregate

    v8f zero = {};
    v8f acc[4][4];
#pragma unroll
    for (int mi = 0; mi < 4; ++mi)
#pragma unroll
        for (int ni = 0; ni < 4; ++ni) acc[mi][ni] = zero;

    // Prologue: fill buffer 0 (A at +0, B at +2*TILE_B).
    issue_tile_pair(gA, gB, DD * 2, lbase, lbase + 2 * TILE_B, tid);

    const int KSTEPS = DD / 32;
    for (int ks = 0; ks < KSTEPS; ++ks) {
        const int cur = ks & 1;
        const unsigned curoff = cur ? TILE_B : 0u;
        const unsigned nxtoff = cur ? 0u : TILE_B;
        if (ks + 1 < KSTEPS) {
            issue_tile_pair(gA + (size_t)(ks + 1) * 64, gB + (size_t)(ks + 1) * 64,
                            DD * 2, lbase + nxtoff, lbase + 2 * TILE_B + nxtoff, tid);
            wait_async<8>();   // 8 newest still in flight => previous batch done
        } else {
            wait_async<0>();
        }
        __syncthreads();

        const u16* sA = smem + (cur ? TILE_LDS : 0);
        const u16* sB = smem + 2 * TILE_LDS + (cur ? TILE_LDS : 0);

        v16bf a[4], b[4];
#pragma unroll
        for (int mi = 0; mi < 4; ++mi)
            a[mi] = load_frag_lds(sA + (wm + mi * 16) * ROWE, lane);
#pragma unroll
        for (int ni = 0; ni < 4; ++ni)
            b[ni] = load_frag_lds(sB + (wn + ni * 16) * ROWE, lane);
#pragma unroll
        for (int mi = 0; mi < 4; ++mi)
#pragma unroll
            for (int ni = 0; ni < 4; ++ni)
                acc[mi][ni] = __builtin_amdgcn_wmma_f32_16x16x32_bf16(
                    false, a[mi], false, b[ni], (short)0, acc[mi][ni], false, false);

        __syncthreads();   // all waves done reading buf[cur] before it is rewritten
        (void)curoff;
    }

    // Epilogue: bias + squared ReLU, store bf16.
    // C layout: VGPR r, lanes 0-15 -> (M=r, N=lane); lanes 16-31 -> (M=r+8).
    const int rsel = lane & 15;
    const int hi   = lane >> 4;
    const int m0   = mb + wm;
    const int n0   = nb + wn;
#pragma unroll
    for (int ni = 0; ni < 4; ++ni) {
        const int ncol   = n0 + ni * 16 + rsel;
        const float bias = b1[(size_t)e * HH + ncol];
#pragma unroll
        for (int mi = 0; mi < 4; ++mi) {
#pragma unroll
            for (int r = 0; r < 8; ++r) {
                const int mrow = m0 + mi * 16 + r + 8 * hi;
                float v = acc[mi][ni][r] + bias;
                v = fmaxf(v, 0.0f);
                v = v * v;
                hsq[((size_t)e * NN + mrow) * HH + ncol] = f2bf(v);
            }
        }
    }
}

// ---------------------------------------------------------------------------
// GEMM2 + combine: out[n,o] = sum_e route[n,e] * (hsq[e,n,:] @ W2[e,o,:] + b2[e,o])
// Same LDS-staged tiling; per-expert accumulator folded into weighted total.
// ---------------------------------------------------------------------------
__global__ __launch_bounds__(128) void gemm2_kernel(const u16* __restrict__ hsq,
                                                    const u16* __restrict__ w2bf,
                                                    const float* __restrict__ b2,
                                                    const float* __restrict__ route,
                                                    float* __restrict__ out) {
    __shared__ __align__(16) u16 smem[4 * TILE_LDS];

    const int tid  = threadIdx.x;
    const int lane = tid & 31;
    const int wid  = tid >> 5;
    const int mb   = blockIdx.y * 128;
    const int nb   = blockIdx.x * 128;
    const int wm   = (wid >> 1) * 64;
    const int wn   = (wid & 1) * 64;
    const int rsel = lane & 15;
    const int hi   = lane >> 4;
    const int m0   = mb + wm;
    const int n0   = nb + wn;

    const unsigned lbase = lds_off_u32(smem);

    v8f zero = {};
    v8f total[4][4];
#pragma unroll
    for (int mi = 0; mi < 4; ++mi)
#pragma unroll
        for (int ni = 0; ni < 4; ++ni) total[mi][ni] = zero;

    for (int e = 0; e < EE; ++e) {
        const char* gA = (const char*)(hsq  + (size_t)e * NN * HH + (size_t)mb * HH);
        const char* gB = (const char*)(w2bf + ((size_t)e * OO + nb) * HH);

        v8f acc[4][4];
#pragma unroll
        for (int mi = 0; mi < 4; ++mi)
#pragma unroll
            for (int ni = 0; ni < 4; ++ni) acc[mi][ni] = zero;

        issue_tile_pair(gA, gB, HH * 2, lbase, lbase + 2 * TILE_B, tid);

        const int KSTEPS = HH / 32;
        for (int ks = 0; ks < KSTEPS; ++ks) {
            const int cur = ks & 1;
            const unsigned nxtoff = cur ? 0u : TILE_B;
            if (ks + 1 < KSTEPS) {
                issue_tile_pair(gA + (size_t)(ks + 1) * 64, gB + (size_t)(ks + 1) * 64,
                                HH * 2, lbase + nxtoff, lbase + 2 * TILE_B + nxtoff, tid);
                wait_async<8>();
            } else {
                wait_async<0>();
            }
            __syncthreads();

            const u16* sA = smem + (cur ? TILE_LDS : 0);
            const u16* sB = smem + 2 * TILE_LDS + (cur ? TILE_LDS : 0);

            v16bf a[4], b[4];
#pragma unroll
            for (int mi = 0; mi < 4; ++mi)
                a[mi] = load_frag_lds(sA + (wm + mi * 16) * ROWE, lane);
#pragma unroll
            for (int ni = 0; ni < 4; ++ni)
                b[ni] = load_frag_lds(sB + (wn + ni * 16) * ROWE, lane);
#pragma unroll
            for (int mi = 0; mi < 4; ++mi)
#pragma unroll
                for (int ni = 0; ni < 4; ++ni)
                    acc[mi][ni] = __builtin_amdgcn_wmma_f32_16x16x32_bf16(
                        false, a[mi], false, b[ni], (short)0, acc[mi][ni], false, false);

            __syncthreads();
        }

        // Fold: total += route[n,e] * (acc + b2[e,o])
#pragma unroll
        for (int mi = 0; mi < 4; ++mi) {
            float wr[8];
#pragma unroll
            for (int r = 0; r < 8; ++r)
                wr[r] = route[(size_t)(m0 + mi * 16 + r + 8 * hi) * EE + e];
#pragma unroll
            for (int ni = 0; ni < 4; ++ni) {
                const float bias = b2[(size_t)e * OO + n0 + ni * 16 + rsel];
#pragma unroll
                for (int r = 0; r < 8; ++r)
                    total[mi][ni][r] += wr[r] * (acc[mi][ni][r] + bias);
            }
        }
    }

#pragma unroll
    for (int mi = 0; mi < 4; ++mi)
#pragma unroll
        for (int ni = 0; ni < 4; ++ni)
#pragma unroll
            for (int r = 0; r < 8; ++r)
                out[(size_t)(m0 + mi * 16 + r + 8 * hi) * OO + (n0 + ni * 16 + rsel)] =
                    total[mi][ni][r];
}

// ---------------------------------------------------------------------------
extern "C" void kernel_launch(void* const* d_in, const int* in_sizes, int n_in,
                              void* d_out, int out_size, void* d_ws, size_t ws_size,
                              hipStream_t stream) {
    const float* x  = (const float*)d_in[0];
    const float* Wr = (const float*)d_in[1];
    const float* br = (const float*)d_in[2];
    const float* W1 = (const float*)d_in[3];
    const float* b1 = (const float*)d_in[4];
    const float* W2 = (const float*)d_in[5];
    const float* b2 = (const float*)d_in[6];
    float* out = (float*)d_out;

    char*  ws  = (char*)d_ws;
    size_t off = 0;
    auto alloc = [&](size_t bytes) -> char* {
        off = (off + 255) & ~(size_t)255;
        char* p = ws + off;
        off += bytes;
        return p;
    };
    float* route = (float*)alloc((size_t)NN * EE * sizeof(float));
    u16*   xbf   = (u16*)  alloc((size_t)NN * DD * sizeof(u16));
    u16*   w1bf  = (u16*)  alloc((size_t)EE * HH * DD * sizeof(u16));
    u16*   w2bf  = (u16*)  alloc((size_t)EE * OO * HH * sizeof(u16));
    u16*   hsq   = (u16*)  alloc((size_t)EE * NN * HH * sizeof(u16));  // 128 MB: L2-resident

    convert_kernel<<<(NN * DD / 4 + 255) / 256, 256, 0, stream>>>(x,  xbf,  NN * DD / 4);
    convert_kernel<<<(EE * HH * DD / 4 + 255) / 256, 256, 0, stream>>>(W1, w1bf, EE * HH * DD / 4);
    convert_kernel<<<(EE * OO * HH / 4 + 255) / 256, 256, 0, stream>>>(W2, w2bf, EE * OO * HH / 4);

    router_kernel<<<NN / 8, 256, 0, stream>>>(x, Wr, br, route);

    gemm1_kernel<<<dim3(HH / 128, NN / 128, EE), 128, 0, stream>>>(xbf, w1bf, b1, hsq);
    gemm2_kernel<<<dim3(OO / 128, NN / 128, 1), 128, 0, stream>>>(hsq, w2bf, b2, route, out);
}
